// MultiHeadAttention_16973710754025
// MI455X (gfx1250) — matrix operand order repaired and checked by reading, not compiled, not run on hardware
//
#include <hip/hip_runtime.h>
#include <hip/hip_bf16.h>

typedef __attribute__((ext_vector_type(16))) __bf16 v16bf;
typedef __attribute__((ext_vector_type(8)))  __bf16 v8bf;
typedef __attribute__((ext_vector_type(8)))  float  v8f;
typedef __attribute__((ext_vector_type(4)))  unsigned int v4u;

typedef int v4i __attribute__((vector_size(16)));
typedef __attribute__((address_space(1))) v4i* as1_v4i;
typedef __attribute__((address_space(3))) v4i* as3_v4i;

#define D_MODEL 1024
#define SEQ     2048
#define BATCH   4
#define NHEADS  16
#define HDIM    64
#define MROWS   (BATCH*SEQ)   /* 8192 */

#if defined(__has_builtin)
#if __has_builtin(__builtin_amdgcn_global_load_async_to_lds_b128)
#define HAS_ASYNC_COPY 1
#endif
#endif
#ifndef HAS_ASYNC_COPY
#define HAS_ASYNC_COPY 0
#endif

#if HAS_ASYNC_COPY
#if __has_builtin(__builtin_amdgcn_s_wait_asynccnt)
#define WAIT_ASYNC() __builtin_amdgcn_s_wait_asynccnt(0)
#else
#define WAIT_ASYNC() asm volatile("s_wait_asynccnt 0x0" ::: "memory")
#endif
#else
#define WAIT_ASYNC() ((void)0)
#endif

__device__ __forceinline__ unsigned short f2bf(float f) {
  unsigned int u = __float_as_uint(f);
  u += 0x7fffu + ((u >> 16) & 1u);       // round-to-nearest-even
  return (unsigned short)(u >> 16);
}

union V16U { v16bf v; v8bf h[2]; };

// 16-byte global -> LDS copy element (async on CDNA5, sync fallback otherwise).
// AS pointers are produced via integer casts: generic LDS pointers carry the
// LDS byte offset in their low 32 bits (ISA 10.2 aperture rules), and global
// addresses are identical in AS1.
__device__ __forceinline__ void copy16_g2l(unsigned short* l, const unsigned short* g) {
#if HAS_ASYNC_COPY
  __builtin_amdgcn_global_load_async_to_lds_b128(
      (as1_v4i)(unsigned long long)g,
      (as3_v4i)(unsigned int)(unsigned long long)l, 0, 0);
#else
  *(v8bf*)l = *(const v8bf*)g;
#endif
}

// ---------------------------------------------------------------- converts
__global__ __launch_bounds__(256) void cvt_f32_to_bf16(const float* __restrict__ in,
                                                       unsigned short* __restrict__ out,
                                                       int n) {
  int i = blockIdx.x * 256 + threadIdx.x;
  if (i < n) out[i] = f2bf(in[i]);
}

// W[K=1024][N=1024] f32 -> Wt[N][K] bf16, optionally pre-scaled
__global__ __launch_bounds__(256) void cvt_transpose_bf16(const float* __restrict__ W,
                                                          unsigned short* __restrict__ Wt,
                                                          float scale) {
  int i = blockIdx.x * 256 + threadIdx.x;   // exactly 1M threads
  int k = i >> 10, n = i & 1023;
  Wt[(size_t)n * D_MODEL + k] = f2bf(W[(size_t)k * D_MODEL + n] * scale);
}

// ---------------------------------------------------------------- GEMM
// C[M=8192, N=1024] = A[M,K=1024](bf16) @ Bt[N,K](bf16)^T + bias*bscale
// MODE 0: bf16 out [M,N]
// MODE 1: bf16 out, transposed per head: Vt[b,h][hd][s]
// MODE 2: f32 out [M,N]
// Wave tile 64x64 (4x4 WMMA), block = 8 waves -> 128(M) x 256(N) per block.
template <int MODE>
__global__ __launch_bounds__(256) void gemm_bf16_wmma(
    const unsigned short* __restrict__ A,
    const unsigned short* __restrict__ Bt,
    const float* __restrict__ bias,
    float bscale,
    void* __restrict__ Cout) {
  const int K = D_MODEL, N = D_MODEL;
  int wid  = threadIdx.x >> 5;
  int lane = threadIdx.x & 31;
  int lm = lane & 15, hi = lane >> 4;
  int m_base = blockIdx.x * 128 + (wid & 1) * 64;   // 2 waves along M
  int n_base = blockIdx.y * 256 + (wid >> 1) * 64;  // 4 waves along N

  v8f acc[4][4] = {};

  for (int k = 0; k < K; k += 32) {
    // prefetch next A k-slab (one cacheline per lane)
    __builtin_prefetch(A + (size_t)(m_base + (lane & 63)) * K + k + 32, 0, 3);
    v16bf afr[4], bfr[4];
#pragma unroll
    for (int ms = 0; ms < 4; ++ms) {
      // A-fragment: lanes 0-15: K 0-15; lanes 16-31: K 16-31
      const unsigned short* ap = A + (size_t)(m_base + ms * 16 + lm) * K + k + hi * 16;
      V16U u;
      u.h[0] = *(const v8bf*)ap;
      u.h[1] = *(const v8bf*)(ap + 8);
      afr[ms] = u.v;
    }
#pragma unroll
    for (int ns = 0; ns < 4; ++ns) {
      // B-fragment: lane column = lm; lanes 0-15 K 0-15, lanes 16-31 K 16-31 (contiguous in Bt)
      const unsigned short* bp = Bt + (size_t)(n_base + ns * 16 + lm) * K + k + hi * 16;
      bfr[ns] = *(const v16bf*)bp;
    }
#pragma unroll
    for (int ms = 0; ms < 4; ++ms)
#pragma unroll
      for (int ns = 0; ns < 4; ++ns)
        acc[ms][ns] = __builtin_amdgcn_wmma_f32_16x16x32_bf16(
            false, afr[ms], false, bfr[ns], (short)0, acc[ms][ns], false, false);
  }

#pragma unroll
  for (int ns = 0; ns < 4; ++ns) {
    int n = n_base + ns * 16 + lm;
    float bv = bias[n] * bscale;
#pragma unroll
    for (int ms = 0; ms < 4; ++ms) {
#pragma unroll
      for (int r = 0; r < 8; ++r) {
        int m = m_base + ms * 16 + r + hi * 8;   // C/D layout
        float val = acc[ms][ns][r] + bv;
        if (MODE == 0) {
          ((unsigned short*)Cout)[(size_t)m * N + n] = f2bf(val);
        } else if (MODE == 1) {
          int b = m >> 11, s = m & (SEQ - 1);
          int h = n >> 6,  d = n & (HDIM - 1);
          ((unsigned short*)Cout)[((size_t)((b * NHEADS + h) * HDIM + d) << 11) + s] = f2bf(val);
        } else {
          ((float*)Cout)[(size_t)m * N + n] = val;
        }
      }
    }
  }
}

// ---------------------------------------------------------------- attention
// Transposed-score flash attention with async double-buffered K/V staging.
// Q,K: [8192, 1024] bf16 (head h = cols h*64..).  Vt: [B*H][64][2048] bf16.
// Out: [8192, 1024] f32.
// Block = 8 waves, one (b,h): K/V chunks staged once per block into LDS via
// GLOBAL_LOAD_ASYNC_TO_LDS_B128 (ASYNCcnt / s_wait_asynccnt), double-buffered
// so the copy of chunk i+1 overlaps compute of chunk i.
__global__ __launch_bounds__(256) void flash_attn_bf16(
    const unsigned short* __restrict__ Q,
    const unsigned short* __restrict__ Km,
    const unsigned short* __restrict__ Vt,
    float* __restrict__ Out) {
  const int Dm = D_MODEL;
  int bh = blockIdx.x;
  int b = bh >> 4, h = bh & 15;
  int tid  = threadIdx.x;
  int wid  = tid >> 5;
  int lane = tid & 31;
  int lm = lane & 15, hi = lane >> 4;
  int qrow0 = blockIdx.y * 128 + wid * 16;

  // K chunk: [32 keys][64 hd]; V chunk: [64 hd][32 keys]; both double-buffered.
  __shared__ __align__(16) unsigned short klds[2][32 * 64];
  __shared__ __align__(16) unsigned short vlds[2][64 * 32];
  // per-wave P^T staging: [query 16][key 32] (wave-private, DS in-order)
  __shared__ __align__(16) unsigned short plds[8][16 * 32];
  unsigned short* pw = plds[wid];

  const unsigned short* kbase = Km + (size_t)b * SEQ * Dm + h * HDIM;
  const unsigned short* vbase = Vt + (size_t)bh * HDIM * SEQ;

  // per-thread copy slots: K = 32 rows x 8 x 16B, V = 64 rows x 4 x 16B
  int krow = tid >> 3, kcol = (tid & 7) * 8;
  int vrow = tid >> 2, vcol = (tid & 3) * 8;

  // B-fragments of Q^T (two 32-wide hd slabs), resident in registers.
  v16bf bq[2];
  const unsigned short* qp0 = Q + (size_t)(b * SEQ + qrow0 + lm) * Dm + h * HDIM + hi * 16;
  bq[0] = *(const v16bf*)(qp0);
  bq[1] = *(const v16bf*)(qp0 + 32);

  v8f o[4] = {};                 // O^T: (hd j*16 + r + 8*hi, query lm)
  float mq = -1e30f, lq = 0.f;   // per-query softmax state (per lane, hi-replicated)

  // prime buffer 0 with chunk kt = 0
  copy16_g2l(&klds[0][krow * 64 + kcol], kbase + (size_t)krow * Dm + kcol);
  copy16_g2l(&vlds[0][vrow * 32 + vcol], vbase + (size_t)vrow * SEQ + vcol);

  for (int kt = 0, it = 0; kt < SEQ; kt += 32, ++it) {
    int cur = it & 1;
    WAIT_ASYNC();         // my async copies (incl. buffer `cur`) have landed in LDS
    __syncthreads();      // everyone's copies landed; prior readers of `cur^1` done
    if (kt + 32 < SEQ) {  // issue next chunk into the other buffer (overlaps compute)
      copy16_g2l(&klds[cur ^ 1][krow * 64 + kcol],
                 kbase + (size_t)(kt + 32 + krow) * Dm + kcol);
      copy16_g2l(&vlds[cur ^ 1][vrow * 32 + vcol],
                 vbase + (size_t)vrow * SEQ + (kt + 32) + vcol);
    }
    const unsigned short* kc = klds[cur];
    const unsigned short* vc = vlds[cur];

    // ---- S^T: two 16(key) x 16(query) tiles; A = K rows (from LDS), B = Q^T
    v8f st0 = {}, st1 = {};
    {
      const unsigned short* ap = kc + lm * 64 + hi * 16;
      V16U u0, u1;
      u0.h[0] = *(const v8bf*)(ap);        u0.h[1] = *(const v8bf*)(ap + 8);
      u1.h[0] = *(const v8bf*)(ap + 32);   u1.h[1] = *(const v8bf*)(ap + 40);
      st0 = __builtin_amdgcn_wmma_f32_16x16x32_bf16(false, u0.v, false, bq[0], (short)0, st0, false, false);
      st0 = __builtin_amdgcn_wmma_f32_16x16x32_bf16(false, u1.v, false, bq[1], (short)0, st0, false, false);
      const unsigned short* ap2 = kc + (16 + lm) * 64 + hi * 16;
      u0.h[0] = *(const v8bf*)(ap2);       u0.h[1] = *(const v8bf*)(ap2 + 8);
      u1.h[0] = *(const v8bf*)(ap2 + 32);  u1.h[1] = *(const v8bf*)(ap2 + 40);
      st1 = __builtin_amdgcn_wmma_f32_16x16x32_bf16(false, u0.v, false, bq[0], (short)0, st1, false, false);
      st1 = __builtin_amdgcn_wmma_f32_16x16x32_bf16(false, u1.v, false, bq[1], (short)0, st1, false, false);
    }

    // ---- online softmax per query (scale folded into Wq/bq)
    float mx = fmaxf(st0[0], st1[0]);
#pragma unroll
    for (int r = 1; r < 8; ++r) mx = fmaxf(mx, fmaxf(st0[r], st1[r]));
    mx = fmaxf(mx, __shfl_xor(mx, 16, 32));      // other 16 keys of this chunk
    float mnew  = fmaxf(mq, mx);
    float alpha = __expf(mq - mnew);
    float rs = 0.f;
#pragma unroll
    for (int r = 0; r < 8; ++r) {
      st0[r] = __expf(st0[r] - mnew);
      st1[r] = __expf(st1[r] - mnew);
      rs += st0[r] + st1[r];
    }
    rs += __shfl_xor(rs, 16, 32);
    lq = lq * alpha + rs;
    mq = mnew;
#pragma unroll
    for (int j = 0; j < 4; ++j)
#pragma unroll
      for (int r = 0; r < 8; ++r) o[j][r] *= alpha;

    // ---- P^T: C-layout -> LDS [query][key], vectorized 16B stores
    v4u pk0, pk1;
#pragma unroll
    for (int i = 0; i < 4; ++i) {
      pk0[i] = (unsigned int)f2bf(st0[2 * i]) | ((unsigned int)f2bf(st0[2 * i + 1]) << 16);
      pk1[i] = (unsigned int)f2bf(st1[2 * i]) | ((unsigned int)f2bf(st1[2 * i + 1]) << 16);
    }
    *(v4u*)(pw + lm * 32 + hi * 8)      = pk0;   // keys  0-15 half owned by this hi
    *(v4u*)(pw + lm * 32 + 16 + hi * 8) = pk1;   // keys 16-31 half
    // B-fragment of P^T: lane = query lm, keys hi*16..+15 (contiguous 32B)
    v16bf pfr = *(const v16bf*)(pw + lm * 32 + hi * 16);

    // ---- O^T += V^T * P^T   (A = V^T rows, from LDS)
#pragma unroll
    for (int j = 0; j < 4; ++j) {
      const unsigned short* vp = vc + (j * 16 + lm) * 32 + hi * 16;
      V16U u;
      u.h[0] = *(const v8bf*)vp;
      u.h[1] = *(const v8bf*)(vp + 8);
      o[j] = __builtin_amdgcn_wmma_f32_16x16x32_bf16(false, u.v, false, pfr, (short)0, o[j], false, false);
    }
  }

  // ---- normalize and write: per lane 8 contiguous f32 per j tile (32B stores)
  float inv = 1.0f / lq;
  float* obase = Out + (size_t)(b * SEQ + qrow0 + lm) * Dm + h * HDIM + hi * 8;
#pragma unroll
  for (int j = 0; j < 4; ++j) {
    v8f ov;
#pragma unroll
    for (int r = 0; r < 8; ++r) ov[r] = o[j][r] * inv;
    *(v8f*)(obase + j * 16) = ov;
  }
}

// ---------------------------------------------------------------- launcher
extern "C" void kernel_launch(void* const* d_in, const int* in_sizes, int n_in,
                              void* d_out, int out_size, void* d_ws, size_t ws_size,
                              hipStream_t stream) {
  const float* x  = (const float*)d_in[0];
  const float* Wq = (const float*)d_in[1];
  const float* bq = (const float*)d_in[2];
  const float* Wk = (const float*)d_in[3];
  const float* bk = (const float*)d_in[4];
  const float* Wv = (const float*)d_in[5];
  const float* bv = (const float*)d_in[6];
  const float* Wo = (const float*)d_in[7];
  const float* bo = (const float*)d_in[8];
  float* out = (float*)d_out;

  const size_t MD = (size_t)MROWS * D_MODEL;         // 8.4M elements
  const size_t WD = (size_t)D_MODEL * D_MODEL;       // 1M elements
  const float ssc = 0.125f;                          // 1/sqrt(HDIM), folded into Q

  char* p = (char*)d_ws;
  unsigned short* xbf    = (unsigned short*)p; p += MD * 2;   // 16 MB
  unsigned short* wq_t   = (unsigned short*)p; p += WD * 2;   //  2 MB
  unsigned short* wk_t   = (unsigned short*)p; p += WD * 2;
  unsigned short* wv_t   = (unsigned short*)p; p += WD * 2;
  unsigned short* wo_t   = (unsigned short*)p; p += WD * 2;
  unsigned short* qbf    = (unsigned short*)p; p += MD * 2;   // 16 MB
  unsigned short* kbf    = (unsigned short*)p; p += MD * 2;   // 16 MB
  unsigned short* vtbf   = (unsigned short*)p; p += MD * 2;   // 16 MB
  float*          attn   = (float*)p;          p += MD * 4;   // 32 MB
  unsigned short* attnbf = (unsigned short*)p; p += MD * 2;   // 16 MB

  cvt_f32_to_bf16<<<(int)(MD / 256), 256, 0, stream>>>(x, xbf, (int)MD);
  cvt_transpose_bf16<<<(int)(WD / 256), 256, 0, stream>>>(Wq, wq_t, ssc);
  cvt_transpose_bf16<<<(int)(WD / 256), 256, 0, stream>>>(Wk, wk_t, 1.0f);
  cvt_transpose_bf16<<<(int)(WD / 256), 256, 0, stream>>>(Wv, wv_t, 1.0f);
  cvt_transpose_bf16<<<(int)(WD / 256), 256, 0, stream>>>(Wo, wo_t, 1.0f);

  dim3 g(MROWS / 128, D_MODEL / 256);
  gemm_bf16_wmma<0><<<g, 256, 0, stream>>>(xbf, wq_t, bq, ssc, qbf);
  gemm_bf16_wmma<0><<<g, 256, 0, stream>>>(xbf, wk_t, bk, 1.0f, kbf);
  gemm_bf16_wmma<1><<<g, 256, 0, stream>>>(xbf, wv_t, bv, 1.0f, vtbf);

  dim3 ga(BATCH * NHEADS, SEQ / 128);
  flash_attn_bf16<<<ga, 256, 0, stream>>>(qbf, kbf, vtbf, attn);

  cvt_f32_to_bf16<<<(int)(MD / 256), 256, 0, stream>>>(attn, attnbf, (int)MD);
  gemm_bf16_wmma<2><<<g, 256, 0, stream>>>(attnbf, wo_t, bo, 1.0f, out);

  (void)in_sizes; (void)n_in; (void)out_size; (void)ws_size;
}
